// DeformableAttention_61881888801375
// MI455X (gfx1250) — compile-verified, matrix operand order checked by reading, not run on hardware
//
#include <hip/hip_runtime.h>
#include <hip/hip_bf16.h>

// ---------------- types for WMMA ----------------
typedef __attribute__((ext_vector_type(16))) __bf16 v16bf;
typedef __attribute__((ext_vector_type(8)))  float  v8f;

// Problem constants (fixed by the reference)
#define NQ_   20000
#define NV_   19560
#define EMB_  256
#define HEADS 4
#define DH    64
#define DDEPTH 64

__device__ __constant__ int c_Hl[4]    = {92, 46, 23, 12};
__device__ __constant__ int c_Wl[4]    = {160, 80, 40, 20};
__device__ __constant__ int c_start[4] = {0, 14720, 18400, 19320};

// packed f32x2 -> bf16x2 (lowers to v_cvt_pk_bf16_f32 on CDNA5)
__device__ __forceinline__ unsigned pack2bf(float lo, float hi) {
  __hip_bfloat162 t = __float22bfloat162_rn(float2{lo, hi});
  union { __hip_bfloat162 b; unsigned u; } x;
  x.b = t;
  return x.u;
}

// ---------------------------------------------------------------------------
// GEMM: C[M,N] = A[M,K=256] * W[K,N] + bias[N] (fp32 memory, bf16 WMMA, f32 acc)
// Block: 256 threads = 8 waves; block tile 64(M) x 256(N); wave tile 32x64.
// Double-buffered LDS staging, one barrier per K-step.
// ---------------------------------------------------------------------------
#define ASTR 40   // LDS row stride (bf16 elems) for A tile (32 + pad)
#define BSTR 40   // LDS row stride for transposed B tile

__device__ __forceinline__ void stageA(unsigned short* __restrict__ As,
                                       const float* __restrict__ A,
                                       int m0, int kk, int M, int tid) {
  int r  = tid >> 2;            // 0..63
  int c0 = (tid & 3) * 8;       // 0,8,16,24
  int row = m0 + r;
  float4 f0 = {0.f, 0.f, 0.f, 0.f}, f1 = {0.f, 0.f, 0.f, 0.f};
  if (row < M) {
    const float* src = A + (long)row * 256 + kk + c0;
    f0 = *(const float4*)(src);
    f1 = *(const float4*)(src + 4);
  }
  unsigned* dst = (unsigned*)&As[r * ASTR + c0];
  dst[0] = pack2bf(f0.x, f0.y);
  dst[1] = pack2bf(f0.z, f0.w);
  dst[2] = pack2bf(f1.x, f1.y);
  dst[3] = pack2bf(f1.z, f1.w);
}

__device__ __forceinline__ void stageB(unsigned short* __restrict__ BsT,
                                       const float* __restrict__ W,
                                       int kk, int N, int tid) {
  int kc = (tid >> 4) * 2;      // 0,2,...,30  (pair of K rows)
  int n0 = (tid & 15) * 16;     // 0..240
  if (n0 >= N) return;
  const float* r0 = W + (long)(kk + kc) * N + n0;
  const float* r1 = r0 + N;
  #pragma unroll
  for (int j4 = 0; j4 < 4; ++j4) {
    float4 a4 = *(const float4*)(r0 + j4 * 4);
    float4 b4 = *(const float4*)(r1 + j4 * 4);
    int nb = n0 + j4 * 4;
    *(unsigned*)&BsT[(nb + 0) * BSTR + kc] = pack2bf(a4.x, b4.x);
    *(unsigned*)&BsT[(nb + 1) * BSTR + kc] = pack2bf(a4.y, b4.y);
    *(unsigned*)&BsT[(nb + 2) * BSTR + kc] = pack2bf(a4.z, b4.z);
    *(unsigned*)&BsT[(nb + 3) * BSTR + kc] = pack2bf(a4.w, b4.w);
  }
}

__global__ __launch_bounds__(256)
void gemm_bf16_wmma(const float* __restrict__ A, const float* __restrict__ W,
                    const float* __restrict__ bias, float* __restrict__ C,
                    int M, int N) {
  const int K = 256, NSTEP = K / 32;
  __shared__ unsigned short As[2][64 * ASTR];    //  64 x 32 bf16 (+pad) x2
  __shared__ unsigned short BsT[2][256 * BSTR];  // 256 x 32 bf16 (+pad) x2

  const int tid  = threadIdx.x;
  const int wave = tid >> 5;
  const int lane = tid & 31;
  const int wm   = wave & 1;       // 0..1
  const int wn   = wave >> 1;      // 0..3
  const int m0   = blockIdx.x * 64;

  v8f acc[2][4] = {};

  const int khalf = lane >> 4;     // 0: K {0..7,16..23}; 1: K {8..15,24..31}
  const int lr    = lane & 15;

  // prologue: stage step 0 into buffer 0
  stageA(As[0], A, m0, 0, M, tid);
  stageB(BsT[0], W, 0, N, tid);

  for (int s = 0; s < NSTEP; ++s) {
    const int cur = s & 1;
    __syncthreads();
    if (s + 1 < NSTEP) {           // issue next tile's loads before compute
      stageA(As[1 - cur], A, m0, (s + 1) * 32, M, tid);
      stageB(BsT[1 - cur], W, (s + 1) * 32, N, tid);
    }

    union Frag { v16bf v; unsigned u[8]; };
    Frag a[2], b[4];
    #pragma unroll
    for (int mi = 0; mi < 2; ++mi) {
      const unsigned short* p = &As[cur][(wm * 32 + mi * 16 + lr) * ASTR];
      #pragma unroll
      for (int j = 0; j < 8; ++j) {
        int kq = ((j & 4) ? 16 : 0) + khalf * 8 + (j & 3) * 2;
        a[mi].u[j] = *(const unsigned*)(p + kq);
      }
    }
    #pragma unroll
    for (int ni = 0; ni < 4; ++ni) {
      const unsigned short* p = &BsT[cur][(wn * 64 + ni * 16 + lr) * BSTR];
      #pragma unroll
      for (int j = 0; j < 8; ++j) {
        int kq = ((j & 4) ? 16 : 0) + khalf * 8 + (j & 3) * 2;
        b[ni].u[j] = *(const unsigned*)(p + kq);
      }
    }
    #pragma unroll
    for (int mi = 0; mi < 2; ++mi)
      #pragma unroll
      for (int ni = 0; ni < 4; ++ni)
        acc[mi][ni] = __builtin_amdgcn_wmma_f32_16x16x32_bf16(
            false, a[mi].v, false, b[ni].v, (short)0, acc[mi][ni], false, false);
  }

  // epilogue: bias add + store (C/D layout: VGPR r -> M = r + 8*khalf)
  #pragma unroll
  for (int ni = 0; ni < 4; ++ni) {
    int col = wn * 64 + ni * 16 + lr;
    if (col >= N) continue;
    float bb = bias[col];
    #pragma unroll
    for (int mi = 0; mi < 2; ++mi) {
      #pragma unroll
      for (int r = 0; r < 8; ++r) {
        int row = m0 + wm * 32 + mi * 16 + khalf * 8 + r;
        if (row < M) C[(long)row * N + col] = acc[mi][ni][r] + bb;
      }
    }
  }
}

// ---------------------------------------------------------------------------
// Softmax over 32 (L*P) per (q, head). One wave32 per (q,h).
// ---------------------------------------------------------------------------
__global__ __launch_bounds__(128)
void softmax32(float* __restrict__ attn) {
  int q = blockIdx.x;
  int h = threadIdx.y;
  int lane = threadIdx.x;
  float* p = attn + (long)q * 128 + h * 32;
  float x = p[lane];
  float m = x;
  #pragma unroll
  for (int s = 16; s; s >>= 1) m = fmaxf(m, __shfl_xor(m, s, 32));
  float e = __expf(x - m);
  float sum = e;
  #pragma unroll
  for (int s = 16; s; s >>= 1) sum += __shfl_xor(sum, s, 32);
  p[lane] = e / sum;
}

// ---------------------------------------------------------------------------
// Sampler: one block (128 thr) per query.
// Phase 1: thread = (h,l,p) -> 4 corner weights (attn * bilinear * depth-interp)
//          + value-row indices into LDS.
// Phase 2: thread = (h, dim-pair) -> 128 float2 gathers from v (L2-resident).
// ---------------------------------------------------------------------------
__global__ __launch_bounds__(128)
void dfa_sample(const float* __restrict__ vproj,   // [NV, 256] (H*DH)
                const float* __restrict__ dpt,     // [NV, 64]
                const float* __restrict__ offuv,   // [NQ, 256]
                const float* __restrict__ offd,    // [NQ, 128]
                const float* __restrict__ attn,    // [NQ, 128]
                const float* __restrict__ refpts,  // [NQ, 4, 3]
                float* __restrict__ out)           // [NQ, 256]
{
  __shared__ float sw[512];
  __shared__ int   srow[512];
  const int q = blockIdx.x;
  const int tid = threadIdx.x;

  // ------------- phase 1 -------------
  {
    int h = tid >> 5;
    int l = (tid >> 3) & 3;
    int p = tid & 7;
    int zi = p & 3;                          // z index = p % Z (Z=4)
    const float* rp = refpts + (long)q * 12 + zi * 3;
    float rx = rp[0], ry = rp[1], rz = rp[2];

    int Hl = c_Hl[l], Wl = c_Wl[l], st = c_start[l];
    int n = q * 128 + h * 32 + l * 8 + p;
    float ou = offuv[2 * n + 0];
    float ov = offuv[2 * n + 1];
    float od = offd[n];
    float aw = attn[n];

    // (ref + off/norm)*dim - 0.5 collapses to ref*dim + off - 0.5
    float x = rx * (float)Wl + ou - 0.5f;
    float y = ry * (float)Hl + ov - 0.5f;
    float z = rz * (float)DDEPTH + od - 0.5f;

    float x0f = floorf(x), y0f = floorf(y), z0f = floorf(z);
    float fx = x - x0f, fy = y - y0f, fz = z - z0f;
    int x0 = (int)x0f, y0 = (int)y0f, z0 = (int)z0f;
    int z0c = min(max(z0, 0), DDEPTH - 1);
    int z1c = min(max(z0 + 1, 0), DDEPTH - 1);
    float wz0 = (1.0f - fz) * ((z0 >= 0 && z0 <= DDEPTH - 1) ? 1.0f : 0.0f);
    float wz1 = fz * ((z0 + 1 >= 0 && z0 + 1 <= DDEPTH - 1) ? 1.0f : 0.0f);

    #pragma unroll
    for (int c = 0; c < 4; ++c) {
      int xi = x0 + (c & 1);
      int yi = y0 + (c >> 1);
      float wx = (c & 1) ? fx : 1.0f - fx;
      float wy = (c >> 1) ? fy : 1.0f - fy;
      float valid = (xi >= 0 && xi < Wl && yi >= 0 && yi < Hl) ? 1.0f : 0.0f;
      int idx = min(max(yi, 0), Hl - 1) * Wl + min(max(xi, 0), Wl - 1);
      int row = st + idx;
      float d0 = dpt[(long)row * DDEPTH + z0c];
      float d1 = dpt[(long)row * DDEPTH + z1c];
      float ds = wz0 * d0 + wz1 * d1;
      sw[tid * 4 + c]   = aw * wx * wy * valid * ds;
      srow[tid * 4 + c] = row;
    }
  }
  __syncthreads();

  // ------------- phase 2 -------------
  int h  = tid >> 5;
  int d2 = (tid & 31) * 2;
  float ax = 0.0f, ay = 0.0f;
  const float* vbase = vproj + h * DH + d2;
  #pragma unroll 4
  for (int i = 0; i < 128; ++i) {
    int s = h * 128 + i;
    float wt = sw[s];
    long row = srow[s];
    const float2 vv = *(const float2*)(vbase + row * EMB_);
    ax = fmaf(wt, vv.x, ax);
    ay = fmaf(wt, vv.y, ay);
  }
  long ob = (long)q * EMB_ + h * DH + d2;
  out[ob]     = ax;
  out[ob + 1] = ay;
}

// ---------------------------------------------------------------------------
extern "C" void kernel_launch(void* const* d_in, const int* in_sizes, int n_in,
                              void* d_out, int out_size, void* d_ws, size_t ws_size,
                              hipStream_t stream) {
  const float* query = (const float*)d_in[0];
  const float* value = (const float*)d_in[1];
  const float* dpt   = (const float*)d_in[2];
  const float* refp  = (const float*)d_in[3];
  // d_in[4], d_in[5]: spatial shapes / level starts (constants, hard-coded)
  const float* Wv  = (const float*)d_in[6];
  const float* bv  = (const float*)d_in[7];
  const float* Wo  = (const float*)d_in[8];
  const float* bo  = (const float*)d_in[9];
  const float* Wod = (const float*)d_in[10];
  const float* bod = (const float*)d_in[11];
  const float* Wa  = (const float*)d_in[12];
  const float* ba  = (const float*)d_in[13];

  float* vbuf = (float*)d_ws;                        // NV * 256
  float* ouv  = vbuf + (long)NV_ * EMB_;             // NQ * 256
  float* odp  = ouv + (long)NQ_ * EMB_;              // NQ * 128
  float* abuf = odp + (long)NQ_ * 128;               // NQ * 128

  dim3 blk(256);
  gemm_bf16_wmma<<<(NV_ + 63) / 64, blk, 0, stream>>>(value, Wv,  bv,  vbuf, NV_, 256);
  gemm_bf16_wmma<<<(NQ_ + 63) / 64, blk, 0, stream>>>(query, Wo,  bo,  ouv,  NQ_, 256);
  gemm_bf16_wmma<<<(NQ_ + 63) / 64, blk, 0, stream>>>(query, Wod, bod, odp,  NQ_, 128);
  gemm_bf16_wmma<<<(NQ_ + 63) / 64, blk, 0, stream>>>(query, Wa,  ba,  abuf, NQ_, 128);

  softmax32<<<NQ_, dim3(32, 4), 0, stream>>>(abuf);

  dfa_sample<<<NQ_, 128, 0, stream>>>(vbuf, dpt, ouv, odp, abuf, refp, (float*)d_out);
}
